// MaskHeadSmallConv_46273977647175
// MI455X (gfx1250) — matrix operand order repaired and checked by reading, not compiled, not run on hardware
//
#include <hip/hip_runtime.h>

typedef __attribute__((ext_vector_type(16))) _Float16 v16h;
typedef __attribute__((ext_vector_type(8)))  _Float16 v8h;
typedef __attribute__((ext_vector_type(8)))  float    v8f;

#define BB   2
#define QQ   16
#define BQ   32
#define N0   4096
#define N1   1024
#define N2   256
#define DIMC 296
#define CTX  288
#define I1C  144
#define I2C  72
#define I3C  36

#define WMMA_F16(A, B, C) \
  __builtin_amdgcn_wmma_f32_16x16x32_f16(false, (A), false, (B), (short)0, (C), false, false)

// Packed activation layout ("B-fragment tiled"):
//   P[(((bq*KS + ks)*NT16 + nt)*32 + lane)*16 + e] = X[k, n]
//   with k = ks*32 + (lane>>4)*16 + e,  n = nt*16 + (lane&15)
__device__ __forceinline__ size_t pidx(int bq, int KS, int NT16, int c, int n) {
  int ks  = c >> 5;
  int hf  = (c >> 4) & 1;
  int e   = c & 15;
  int nt  = n >> 4;
  int l16 = n & 15;
  return ((((size_t)bq * KS + ks) * NT16 + nt) * 32 + (hf * 16 + l16)) * 16 + e;
}

// ---------------------------------------------------------------------------
// Fold conv-bias + BatchNorm(eval) into per-channel scale/bias, zero-padded
// to Op rows so the GEMM epilogue writes zeros into pad rows for free.
// ---------------------------------------------------------------------------
__global__ void prep_sb_kernel(const float* __restrict__ bn, const float* __restrict__ bconv,
                               float* __restrict__ scale, float* __restrict__ bias,
                               int C, int Op) {
  int i = blockIdx.x * blockDim.x + threadIdx.x;
  if (i >= Op) return;
  if (i < C) {
    float g = bn[i], be = bn[C + i], m = bn[2 * C + i], v = bn[3 * C + i];
    float s = g * rsqrtf(v + 1e-5f);
    scale[i] = s;
    bias[i]  = (bconv[i] - m) * s + be;
  } else {
    scale[i] = 0.0f;
    bias[i]  = 0.0f;
  }
}

// ---------------------------------------------------------------------------
// Pack fp32 weights (O,C) into f16 WMMA A-fragments, padded to (Op, Cp):
// element ((tm*KS + ks)*32 + lane)*16 + e holds W[m,k] with
//   m = tm*16 + lane%16,  k = ks*32 + (lane>>4)*16 + e
// contiguous k order, the same as the packed B fragments. One contiguous v16h per lane.
// ---------------------------------------------------------------------------
__global__ void pack_w_kernel(const float* __restrict__ W, _Float16* __restrict__ Wp,
                              int O, int C, int Op, int Cp) {
  int t = blockIdx.x * blockDim.x + threadIdx.x;
  int total = Op * Cp;
  if (t >= total) return;
  int KS   = Cp >> 5;
  int e    = t & 15;
  int lane = (t >> 4) & 31;
  int rest = t >> 9;
  int ks   = rest % KS;
  int tm   = rest / KS;
  int l16 = lane & 15, hf = lane >> 4;
  int v = e >> 1, lo = e & 1;
  int k = ks * 32 + hf * 16 + e;
  int m = tm * 16 + l16;
  float val = (m < O && k < C) ? W[(size_t)m * C + k] : 0.0f;
  Wp[t] = (_Float16)val;
}

// h0 = concat([expand(x,Q), bbox_mask]) -> packed layout, 320 rows (zero pad)
__global__ void assemble_h0_kernel(const float* __restrict__ x, const float* __restrict__ bbox,
                                   _Float16* __restrict__ h0) {
  const int KS = 10, NT16 = 16; // 320 rows, 256 cols
  size_t t = (size_t)blockIdx.x * blockDim.x + threadIdx.x;
  const size_t total = (size_t)BQ * KS * NT16 * 512;
  if (t >= total) return;
  int e    = (int)(t & 15);
  int lane = (int)((t >> 4) & 31);
  size_t r = t >> 9;
  int nt = (int)(r % NT16); r /= NT16;
  int ks = (int)(r % KS);
  int bq = (int)(r / KS);
  int c = ks * 32 + (lane >> 4) * 16 + e;
  int n = nt * 16 + (lane & 15);
  int b = bq / QQ, q = bq % QQ;
  float v = 0.0f;
  if (c < CTX)       v = x[((size_t)b * CTX + c) * N2 + n];
  else if (c < DIMC) v = bbox[(((size_t)b * QQ + q) * 8 + (c - CTX)) * N2 + n];
  h0[t] = (_Float16)v;
}

// ---------------------------------------------------------------------------
// WMMA GEMM, fused BN/ReLU epilogue. Software-pipelined, branch-free K loop:
//   while computing K-step k (4 x v_wmma, shared A), the loads for K-step k+1
//   (1 A v16h + 4 B v16h, all contiguous b128 pairs) are already in flight.
// Epilogue: 4 x 16-byte stores into the packed layout of Y.
// Op, Cp multiples of 32; scale/bias zero-padded to Op.
// ---------------------------------------------------------------------------
__global__ void gemm_bn_relu_kernel(const _Float16* __restrict__ Wp,  // packed A fragments
                                    const float* __restrict__ scale,  // (Op)
                                    const float* __restrict__ bias,   // (Op)
                                    const _Float16* __restrict__ X,   // packed (BQ,KS,NT16,32,16)
                                    _Float16* __restrict__ Y,         // packed (BQ,KSy,NT16,32,16)
                                    int Op, int Cp, int NT16) {
  const int lane = threadIdx.x;   // wave32
  const int hf   = lane >> 4;
  const int l16  = lane & 15;
  const int tm   = blockIdx.y;
  const int nt0  = blockIdx.x * 4;
  const int bq   = blockIdx.z;
  const int KS   = Cp >> 5;

  const _Float16* ap = Wp + ((size_t)tm * KS * 32 + lane) * 16;
  const _Float16* xp = X + (((size_t)bq * KS * NT16 + nt0) * 32 + lane) * 16;
  const size_t xstep = (size_t)NT16 * 512;

  // prologue: K-step 0 fragments
  v16h a  = *(const v16h*)ap;
  v16h b0 = *(const v16h*)(xp);
  v16h b1 = *(const v16h*)(xp + 512);
  v16h b2 = *(const v16h*)(xp + 1024);
  v16h b3 = *(const v16h*)(xp + 1536);

  v8f acc0 = {}, acc1 = {}, acc2 = {}, acc3 = {};

  for (int ks = 0; ks < KS - 1; ++ks) {
    ap += 512;
    xp += xstep;
    // prefetch next K-step while current WMMAs run
    v16h an = *(const v16h*)ap;
    v16h c0 = *(const v16h*)(xp);
    v16h c1 = *(const v16h*)(xp + 512);
    v16h c2 = *(const v16h*)(xp + 1024);
    v16h c3 = *(const v16h*)(xp + 1536);
    acc0 = WMMA_F16(a, b0, acc0);
    acc1 = WMMA_F16(a, b1, acc1);
    acc2 = WMMA_F16(a, b2, acc2);
    acc3 = WMMA_F16(a, b3, acc3);
    a = an; b0 = c0; b1 = c1; b2 = c2; b3 = c3;
  }
  acc0 = WMMA_F16(a, b0, acc0);
  acc1 = WMMA_F16(a, b1, acc1);
  acc2 = WMMA_F16(a, b2, acc2);
  acc3 = WMMA_F16(a, b3, acc3);

  // Epilogue: scale/bias + ReLU, vectorized 16B stores into packed Y
  const int mbase = tm * 16 + hf * 8;
  float sc[8], bi[8];
#pragma unroll
  for (int r = 0; r < 8; ++r) { sc[r] = scale[mbase + r]; bi[r] = bias[mbase + r]; }
  const int KSy = Op >> 5;
  _Float16* yb = Y + ((((size_t)bq * KSy + (tm >> 1)) * NT16 + nt0) * 32 +
                      ((tm & 1) * 16 + l16)) * 16 + hf * 8;
  v8f accs[4] = {acc0, acc1, acc2, acc3};
#pragma unroll
  for (int t = 0; t < 4; ++t) {
    v8h o;
#pragma unroll
    for (int r = 0; r < 8; ++r)
      o[r] = (_Float16)fmaxf(accs[t][r] * sc[r] + bi[r], 0.0f);
    *(v8h*)(yb + t * 512) = o;
  }
}

// Adapter 1x1 conv (tiny K; no BN/ReLU), per true batch (B=2), row-major f16 out.
__global__ void adapter_kernel(const float* __restrict__ W, const float* __restrict__ bias,
                               const float* __restrict__ X, _Float16* __restrict__ Y,
                               int O, int C, int N) {
  size_t t = (size_t)blockIdx.x * blockDim.x + threadIdx.x;
  size_t total = (size_t)BB * O * N;
  if (t >= total) return;
  int n = (int)(t % N);
  size_t r = t / N;
  int o = (int)(r % O);
  int b = (int)(r / O);
  const float* Xp = X + (size_t)b * C * N + n;
  const float* Wp = W + (size_t)o * C;
  float acc = bias[o];
  for (int c = 0; c < C; ++c) acc += Wp[c] * Xp[(size_t)c * N];
  Y[t] = (_Float16)acc;
}

// three_nn per true batch (xyz identical across q): 3 smallest d2 + inv-dist weights
__global__ void three_nn_kernel(const float* __restrict__ uxyz, const float* __restrict__ kxyz,
                                int Nu, int Nk, float* __restrict__ w, int* __restrict__ idx) {
  int t = blockIdx.x * blockDim.x + threadIdx.x;
  if (t >= BB * Nu) return;
  int b = t / Nu, u = t % Nu;
  const float* up = uxyz + ((size_t)b * Nu + u) * 3;
  float ux = up[0], uy = up[1], uz = up[2];
  const float* kp = kxyz + (size_t)b * Nk * 3;
  float d0 = 1e30f, d1 = 1e30f, d2 = 1e30f;
  int i0 = 0, i1 = 0, i2 = 0;
  for (int j = 0; j < Nk; ++j) {
    float dx = ux - kp[3 * j], dy = uy - kp[3 * j + 1], dz = uz - kp[3 * j + 2];
    float d = dx * dx + dy * dy + dz * dz;
    if (d < d0)      { d2 = d1; i2 = i1; d1 = d0; i1 = i0; d0 = d; i0 = j; }
    else if (d < d1) { d2 = d1; i2 = i1; d1 = d;  i1 = j; }
    else if (d < d2) { d2 = d;  i2 = j; }
  }
  float w0 = 1.0f / (d0 + 1e-8f), w1 = 1.0f / (d1 + 1e-8f), w2 = 1.0f / (d2 + 1e-8f);
  float s = 1.0f / (w0 + w1 + w2);
  float* wp = w + 3 * (size_t)t;
  int*   ip = idx + 3 * (size_t)t;
  wp[0] = w0 * s; wp[1] = w1 * s; wp[2] = w2 * s;
  ip[0] = i0; ip[1] = i1; ip[2] = i2;
}

// Build packed concat input G (KSg*32 rows):
//   rows [0,Ci): 3-NN interp of packed H;  [Ci,Ci+Cf): row-major F;  rest: 0
__global__ void build_concat_kernel(const _Float16* __restrict__ H, // packed (BQ,KSh,NTh,32,16)
                                    const _Float16* __restrict__ F, // row-major (B, Cf, Nu)
                                    const float* __restrict__ w, const int* __restrict__ idx,
                                    _Float16* __restrict__ G,       // packed (BQ,KSg,Nu/16,32,16)
                                    int Ci, int Cf, int KSg, int KSh, int NTh, int Nu) {
  const int NTg = Nu >> 4;
  size_t t = (size_t)blockIdx.x * blockDim.x + threadIdx.x;
  size_t total = (size_t)BQ * KSg * NTg * 512;
  if (t >= total) return;
  int e    = (int)(t & 15);
  int lane = (int)((t >> 4) & 31);
  size_t r = t >> 9;
  int nt = (int)(r % NTg); r /= NTg;
  int ks = (int)(r % KSg);
  int bq = (int)(r / KSg);
  int c = ks * 32 + (lane >> 4) * 16 + e;
  int u = nt * 16 + (lane & 15);
  int b = bq / QQ;
  float val = 0.0f;
  if (c < Ci) {
    const float* wp = w + ((size_t)b * Nu + u) * 3;
    const int*   ip = idx + ((size_t)b * Nu + u) * 3;
    val = wp[0] * (float)H[pidx(bq, KSh, NTh, c, ip[0])] +
          wp[1] * (float)H[pidx(bq, KSh, NTh, c, ip[1])] +
          wp[2] * (float)H[pidx(bq, KSh, NTh, c, ip[2])];
  } else if (c < Ci + Cf) {
    val = (float)F[((size_t)b * Cf + (c - Ci)) * Nu + u];
  }
  G[t] = (_Float16)val;
}

// Output head: out[bq, n] = out_b + sum_c out_w[c] * h6[bq, c, n]  (h6 packed, fp32 out)
__global__ void out_kernel(const float* __restrict__ W, const float* __restrict__ b,
                           const _Float16* __restrict__ H, float* __restrict__ out) {
  const int KSh = 2, NTh = 256; // 64 rows x 4096 cols, packed
  size_t t = (size_t)blockIdx.x * blockDim.x + threadIdx.x;
  size_t total = (size_t)BQ * N0;
  if (t >= total) return;
  int n  = (int)(t % N0);
  int bq = (int)(t / N0);
  float acc = b[0];
#pragma unroll
  for (int c = 0; c < I3C; ++c) acc += W[c] * (float)H[pidx(bq, KSh, NTh, c, n)];
  out[t] = acc;
}

// ---------------------------------------------------------------------------
extern "C" void kernel_launch(void* const* d_in, const int* in_sizes, int n_in,
                              void* d_out, int out_size, void* d_ws, size_t ws_size,
                              hipStream_t stream) {
  (void)in_sizes; (void)n_in; (void)out_size; (void)ws_size;
  const float* x      = (const float*)d_in[0];
  const float* bbox   = (const float*)d_in[1];
  const float* sa0f   = (const float*)d_in[2];
  const float* sa1f   = (const float*)d_in[3];
  const float* sa0xyz = (const float*)d_in[4];
  const float* sa1xyz = (const float*)d_in[5];
  const float* sa2xyz = (const float*)d_in[6];
  const float* w1   = (const float*)d_in[7];  const float* b1   = (const float*)d_in[8];  const float* bn1   = (const float*)d_in[9];
  const float* w2   = (const float*)d_in[10]; const float* b2   = (const float*)d_in[11]; const float* bn2   = (const float*)d_in[12];
  const float* a1w  = (const float*)d_in[13]; const float* a1b  = (const float*)d_in[14];
  const float* a2w  = (const float*)d_in[15]; const float* a2b  = (const float*)d_in[16];
  const float* p1w1 = (const float*)d_in[17]; const float* p1b1 = (const float*)d_in[18]; const float* p1bn1 = (const float*)d_in[19];
  const float* p1w2 = (const float*)d_in[20]; const float* p1b2 = (const float*)d_in[21]; const float* p1bn2 = (const float*)d_in[22];
  const float* p2w1 = (const float*)d_in[23]; const float* p2b1 = (const float*)d_in[24]; const float* p2bn1 = (const float*)d_in[25];
  const float* p2w2 = (const float*)d_in[26]; const float* p2b2 = (const float*)d_in[27]; const float* p2bn2 = (const float*)d_in[28];
  const float* outw = (const float*)d_in[29]; const float* outb = (const float*)d_in[30];

  // Workspace layout (lifetime-aliased regions, 256B aligned)
  char* ws = (char*)d_ws;
  size_t off = 0;
  auto take = [&](size_t bytes) -> char* {
    char* p = ws + off;
    off += (bytes + 255) & ~(size_t)255;
    return p;
  };
  // packed row counts (multiples of 32):
  //   h0/h1: 320x256   h2: 160x256   g1: 288x1024  h3: 160x1024
  //   h4: 96x1024      g2: 160x4096  h5: 96x4096   h6: 64x4096
  _Float16* R0 = (_Float16*)take((size_t)BQ * 160 * N0 * 2); // h0, later g2
  _Float16* R1 = (_Float16*)take((size_t)BQ * 96  * N0 * 2); // h1, later h5
  _Float16* R2 = (_Float16*)take((size_t)BQ * 64  * N0 * 2); // h2, later h6
  _Float16* g1 = (_Float16*)take((size_t)BQ * 288 * N1 * 2);
  _Float16* h3 = (_Float16*)take((size_t)BQ * 160 * N1 * 2);
  _Float16* h4 = (_Float16*)take((size_t)BQ * 96  * N1 * 2);
  _Float16* f1 = (_Float16*)take((size_t)BB * I1C * N1 * 2);
  _Float16* f0 = (_Float16*)take((size_t)BB * I2C * N0 * 2);
  float* nn1w = (float*)take((size_t)BB * N1 * 3 * 4);
  int*   nn1i = (int*)  take((size_t)BB * N1 * 3 * 4);
  float* nn0w = (float*)take((size_t)BB * N0 * 3 * 4);
  int*   nn0i = (int*)  take((size_t)BB * N0 * 3 * 4);
  // packed f16 weights (fragment order), Op x Cp
  _Float16* pw1  = (_Float16*)take((size_t)320 * 320 * 2);
  _Float16* pw2  = (_Float16*)take((size_t)160 * 320 * 2);
  _Float16* pp11 = (_Float16*)take((size_t)160 * 288 * 2);
  _Float16* pp12 = (_Float16*)take((size_t)96  * 160 * 2);
  _Float16* pp21 = (_Float16*)take((size_t)96  * 160 * 2);
  _Float16* pp22 = (_Float16*)take((size_t)64  * 96  * 2);
  float* s1 = (float*)take(320 * 4); float* t1 = (float*)take(320 * 4);
  float* s2 = (float*)take(160 * 4); float* t2 = (float*)take(160 * 4);
  float* s3 = (float*)take(160 * 4); float* t3 = (float*)take(160 * 4);
  float* s4 = (float*)take(96  * 4); float* t4 = (float*)take(96  * 4);
  float* s5 = (float*)take(96  * 4); float* t5 = (float*)take(96  * 4);
  float* s6 = (float*)take(64  * 4); float* t6 = (float*)take(64  * 4);

  _Float16* h0 = R0; _Float16* g2 = R0;
  _Float16* h1 = R1; _Float16* h5 = R1;
  _Float16* h2 = R2; _Float16* h6 = R2;

  auto prep = [&](const float* bn, const float* bc, float* s, float* t, int C, int Op) {
    prep_sb_kernel<<<dim3((Op + 63) / 64), dim3(64), 0, stream>>>(bn, bc, s, t, C, Op);
  };
  auto pack = [&](const float* W, _Float16* Wp, int O, int C, int Op, int Cp) {
    int total = Op * Cp;
    pack_w_kernel<<<dim3((total + 255) / 256), dim3(256), 0, stream>>>(W, Wp, O, C, Op, Cp);
  };
  auto gemm = [&](const _Float16* Wp, const float* s, const float* t,
                  const _Float16* X, _Float16* Y, int Op, int Cp, int N) {
    int NT16 = N >> 4;
    dim3 grid(NT16 / 4, Op / 16, BQ);
    gemm_bn_relu_kernel<<<grid, dim3(32), 0, stream>>>(Wp, s, t, X, Y, Op, Cp, NT16);
  };

  // BN folding + weight packing (tiny)
  prep(bn1, b1, s1, t1, DIMC, 320);
  prep(bn2, b2, s2, t2, I1C, 160);
  prep(p1bn1, p1b1, s3, t3, I1C, 160);
  prep(p1bn2, p1b2, s4, t4, I2C, 96);
  prep(p2bn1, p2b1, s5, t5, I2C, 96);
  prep(p2bn2, p2b2, s6, t6, I3C, 64);
  pack(w1,   pw1,  DIMC, DIMC,    320, 320);
  pack(w2,   pw2,  I1C,  DIMC,    160, 320);
  pack(p1w1, pp11, I1C,  2 * I1C, 160, 288);
  pack(p1w2, pp12, I2C,  I1C,     96,  160);
  pack(p2w1, pp21, I2C,  2 * I2C, 96,  160);
  pack(p2w2, pp22, I3C,  I2C,     64,  96);

  // Stage 0: assemble packed concat input, layers 1/2 on N2=256
  {
    size_t tot = (size_t)BQ * 320 * N2;
    assemble_h0_kernel<<<dim3((unsigned)((tot + 255) / 256)), dim3(256), 0, stream>>>(x, bbox, h0);
  }
  gemm(pw1, s1, t1, h0, h1, 320, 320, N2);
  gemm(pw2, s2, t2, h1, h2, 160, 320, N2);

  // Adapters (per true batch, row-major outputs)
  {
    size_t tot0 = (size_t)BB * I2C * N0;
    adapter_kernel<<<dim3((unsigned)((tot0 + 255) / 256)), dim3(256), 0, stream>>>(a1w, a1b, sa0f, f0, I2C, 3, N0);
    size_t tot1 = (size_t)BB * I1C * N1;
    adapter_kernel<<<dim3((unsigned)((tot1 + 255) / 256)), dim3(256), 0, stream>>>(a2w, a2b, sa1f, f1, I1C, 128, N1);
  }

  // NN search (per true batch; xyz identical across q)
  three_nn_kernel<<<dim3((BB * N1 + 127) / 128), dim3(128), 0, stream>>>(sa1xyz, sa2xyz, N1, N2, nn1w, nn1i);
  three_nn_kernel<<<dim3((BB * N0 + 127) / 128), dim3(128), 0, stream>>>(sa0xyz, sa1xyz, N0, N1, nn0w, nn0i);

  // FP1: interp(sa2->sa1) + concat (288 rows), two WMMA layers on N1=1024
  {
    size_t tot = (size_t)BQ * 288 * N1;
    build_concat_kernel<<<dim3((unsigned)((tot + 255) / 256)), dim3(256), 0, stream>>>(
        h2, f1, nn1w, nn1i, g1, I1C, I1C, /*KSg=*/9, /*KSh=*/5, /*NTh=*/16, N1);
  }
  gemm(pp11, s3, t3, g1, h3, 160, 288, N1);
  gemm(pp12, s4, t4, h3, h4, 96, 160, N1);

  // FP2: interp(sa1->sa0) + concat (144 rows + pad to 160), on N0=4096
  {
    size_t tot = (size_t)BQ * 160 * N0;
    build_concat_kernel<<<dim3((unsigned)((tot + 255) / 256)), dim3(256), 0, stream>>>(
        h4, f0, nn0w, nn0i, g2, I2C, I2C, /*KSg=*/5, /*KSh=*/3, /*NTh=*/64, N0);
  }
  gemm(pp21, s5, t5, g2, h5, 96, 160, N0);
  gemm(pp22, s6, t6, h5, h6, 64, 96, N0);

  // Output head (fp32)
  {
    size_t tot = (size_t)BQ * N0;
    out_kernel<<<dim3((unsigned)((tot + 255) / 256)), dim3(256), 0, stream>>>(outw, outb, h6, (float*)d_out);
  }
}
